// HyperConvLayer_42502996361359
// MI455X (gfx1250) — matrix operand order repaired and checked by reading, compiled, not run on hardware
//
#include <hip/hip_runtime.h>
#include <cstdint>

// ---------------------------------------------------------------------------
// HyperConv layer for MI455X (gfx1250): bf16 WMMA GEMM pipeline, fp32 accum.
// b=16, n=e=2048, F_IN=F_OUT=256 (hardcoded per reference).
// ---------------------------------------------------------------------------

typedef __attribute__((ext_vector_type(16))) __bf16 v16bf;
typedef __attribute__((ext_vector_type(8)))  float  v8f;

#define BATCH   16
#define NN      2048       // nodes
#define NE      2048       // edges
#define FD      256        // feature dim
#define LDA     40         // LDS row stride in ushorts (32 data + 8 pad)

__device__ __forceinline__ unsigned short f2bf(float f) {
  unsigned int u = __float_as_uint(f);
  u += 0x7FFFu + ((u >> 16) & 1u);            // round-to-nearest-even
  return (unsigned short)(u >> 16);
}

union FragBF { v16bf v; uint4 q[2]; };
union U16x8  { uint4 q; unsigned short s[8]; };

// ---------------------------------------------------------------------------
// Pass 1: row sums (over edges, per node) and column sums (over nodes, per
// edge) of H in a single sweep. Row-sum inverses written directly; column
// sums accumulated with atomics and inverted by finalize_inv.
// grid = (16 row-blocks, 16 batches), block = 256
// ---------------------------------------------------------------------------
__global__ __launch_bounds__(256) void sums_kernel(const float* __restrict__ H,
                                                   float* __restrict__ colsum,
                                                   float* __restrict__ rowInv) {
  const int b  = blockIdx.y;
  const int r0 = blockIdx.x * 128;
  const float* Hb = H + (size_t)b * NN * NE;
  __shared__ float red[256];
  float colAcc[8];
#pragma unroll
  for (int j = 0; j < 8; ++j) colAcc[j] = 0.f;

  for (int r = 0; r < 128; ++r) {
    const float* row = Hb + (size_t)(r0 + r) * NE;
    float rp = 0.f;
#pragma unroll
    for (int j = 0; j < 8; ++j) {
      float v = row[threadIdx.x + 256 * j];
      colAcc[j] += v;
      rp += v;
    }
    red[threadIdx.x] = rp;
    __syncthreads();
    for (int s = 128; s > 0; s >>= 1) {
      if ((int)threadIdx.x < s) red[threadIdx.x] += red[threadIdx.x + s];
      __syncthreads();
    }
    if (threadIdx.x == 0) {
      float s = red[0];
      rowInv[b * NN + r0 + r] = (s > 0.f) ? (1.f / s) : 0.f;
    }
    __syncthreads();
  }
#pragma unroll
  for (int j = 0; j < 8; ++j)
    atomicAdd(&colsum[b * NE + threadIdx.x + 256 * j], colAcc[j]);
}

__global__ void finalize_inv(const float* __restrict__ s,
                             float* __restrict__ inv, int n) {
  int i = blockIdx.x * 256 + threadIdx.x;
  if (i < n) {
    float v = s[i];
    inv[i] = (v > 0.f) ? (1.f / v) : 0.f;
  }
}

// ---------------------------------------------------------------------------
// Batched H-GEMM:  Out[b][m][f] = rowInv[b][m] * sum_k A[b][m][k] * B[b][k][f]
//   TRANS_A = false: A[m][k] = H[b][m][k]        (E_to_x path, m = node)
//   TRANS_A = true : A[m][k] = H[b][k][m]        (x_to_E path, m = edge)
// H fp32 -> bf16 in the LDS stage; B already bf16 (K x 256, row major).
// Block tile 128(M) x 256(N), 8 waves (4x2), wave tile 32 x 128 (2x8 WMMA).
// grid = (16 M-tiles, 16 batches), block = 256
// ---------------------------------------------------------------------------
template <bool TRANS_A>
__global__ __launch_bounds__(256) void h_gemm(const float* __restrict__ H,
                                              const unsigned short* __restrict__ Bsrc,
                                              const float* __restrict__ rowInv,
                                              unsigned short* __restrict__ Out) {
  const int b  = blockIdx.y;
  const int m0 = blockIdx.x * 128;
  const float* Hb = H + (size_t)b * NN * NE;
  const unsigned short* Bb = Bsrc + (size_t)b * 2048 * FD;

  __shared__ __align__(16) unsigned short lA[128 * LDA];  // [m][k]
  __shared__ __align__(16) unsigned short lB[256 * LDA];  // [n][k] (B col-major)

  const int tid  = threadIdx.x;
  const int lane = tid & 31;
  const int wid  = tid >> 5;
  const int wm   = wid & 3;     // 4 waves along M (32 rows each)
  const int wn   = wid >> 2;    // 2 waves along N (128 cols each)
  const int hx   = lane >> 4;   // lane half
  const int l16  = lane & 15;

  v8f acc[2][8];
#pragma unroll
  for (int mt = 0; mt < 2; ++mt)
#pragma unroll
    for (int nt = 0; nt < 8; ++nt) acc[mt][nt] = {};

  for (int k0 = 0; k0 < 2048; k0 += 32) {
    // ---- stage A tile (128 x 32) into LDS as bf16 ----
    if (TRANS_A) {
#pragma unroll
      for (int p = 0; p < 4; ++p) {
        int k = p * 8 + (tid >> 5);         // 0..31 (row of H)
        int c = (tid & 31) * 4;             // 0..124 (col of H = m)
        const float4 v = *(const float4*)(Hb + (size_t)(k0 + k) * NE + m0 + c);
        lA[(c + 0) * LDA + k] = f2bf(v.x);
        lA[(c + 1) * LDA + k] = f2bf(v.y);
        lA[(c + 2) * LDA + k] = f2bf(v.z);
        lA[(c + 3) * LDA + k] = f2bf(v.w);
      }
    } else {
#pragma unroll
      for (int p = 0; p < 4; ++p) {
        int m  = p * 32 + (tid >> 3);       // 0..127
        int cg = (tid & 7) * 4;             // 0..28
        const float4 v = *(const float4*)(Hb + (size_t)(m0 + m) * NE + k0 + cg);
        int o = m * LDA + cg;
        lA[o + 0] = f2bf(v.x); lA[o + 1] = f2bf(v.y);
        lA[o + 2] = f2bf(v.z); lA[o + 3] = f2bf(v.w);
      }
    }
    // ---- stage B tile (32 x 256) bf16, transposed to [n][k] ----
    {
      int k  = tid >> 3;                    // 0..31
      int c0 = (tid & 7) * 32;              // 0..224
      const unsigned short* src = Bb + (size_t)(k0 + k) * FD + c0;
#pragma unroll
      for (int q = 0; q < 4; ++q) {
        U16x8 u; u.q = *(const uint4*)(src + q * 8);
#pragma unroll
        for (int j = 0; j < 8; ++j) lB[(c0 + q * 8 + j) * LDA + k] = u.s[j];
      }
    }
    __syncthreads();

    // ---- fragments (ISA 16-bit A/B layouts) + 16 WMMAs ----
    FragBF a[2], bfr[8];
#pragma unroll
    for (int mt = 0; mt < 2; ++mt) {
      int aOff = (wm * 32 + mt * 16 + l16) * LDA;
      a[mt].q[0] = *(const uint4*)&lA[aOff + 8 * hx];        // K 0..7  (+8*half)
      a[mt].q[1] = *(const uint4*)&lA[aOff + 16 + 8 * hx];   // K 16..23(+8*half)
    }
#pragma unroll
    for (int nt = 0; nt < 8; ++nt) {
      int bOff = (wn * 128 + nt * 16 + l16) * LDA;
      bfr[nt].q[0] = *(const uint4*)&lB[bOff + 8 * hx];      // K 0..7  (+8*half)
      bfr[nt].q[1] = *(const uint4*)&lB[bOff + 16 + 8 * hx]; // K 16..23(+8*half)
    }
#pragma unroll
    for (int mt = 0; mt < 2; ++mt)
#pragma unroll
      for (int nt = 0; nt < 8; ++nt)
        acc[mt][nt] = __builtin_amdgcn_wmma_f32_16x16x32_bf16(
            false, a[mt].v, false, bfr[nt].v, (short)0, acc[mt][nt], false, false);
    __syncthreads();
  }

  // ---- epilogue: per-row normalization scale, store bf16 ----
  const float* sInv = rowInv + b * 2048 + m0;
  unsigned short* Ob = Out + (size_t)b * 2048 * FD;
#pragma unroll
  for (int mt = 0; mt < 2; ++mt) {
#pragma unroll
    for (int r = 0; r < 8; ++r) {
      int ml = wm * 32 + mt * 16 + r + 8 * hx;
      float sc = sInv[ml];
#pragma unroll
      for (int nt = 0; nt < 8; ++nt) {
        int gn = wn * 128 + nt * 16 + l16;
        Ob[(size_t)(m0 + ml) * FD + gn] = f2bf(acc[mt][nt][r] * sc);
      }
    }
  }
}

// ---------------------------------------------------------------------------
// Linear GEMM: Out = relu( concat(A0_bf16[:,0:256], A1_f32[:,0:256]) * W + b )
// M = 32768 (batch folded), N = 256, K = KTOT (256 or 512). W fp32 -> bf16 in
// LDS. Writes fp32 (to d_out) and/or bf16 (for the next stage).
// grid = (256 M-tiles), block = 256
// ---------------------------------------------------------------------------
template <bool HAS_A0, bool OUT_F32, bool OUT_BF16, int KTOT>
__global__ __launch_bounds__(256) void lin_gemm(const unsigned short* __restrict__ A0,
                                                const float* __restrict__ A1,
                                                const float* __restrict__ W,
                                                const float* __restrict__ bias,
                                                float* __restrict__ OutF,
                                                unsigned short* __restrict__ OutBf) {
  constexpr int K0 = HAS_A0 ? 256 : 0;
  const int m0 = blockIdx.x * 128;

  __shared__ __align__(16) unsigned short lA[128 * LDA];
  __shared__ __align__(16) unsigned short lB[256 * LDA];

  const int tid  = threadIdx.x;
  const int lane = tid & 31;
  const int wid  = tid >> 5;
  const int wm   = wid & 3;
  const int wn   = wid >> 2;
  const int hx   = lane >> 4;
  const int l16  = lane & 15;

  v8f acc[2][8];
#pragma unroll
  for (int mt = 0; mt < 2; ++mt)
#pragma unroll
    for (int nt = 0; nt < 8; ++nt) acc[mt][nt] = {};

  for (int k0 = 0; k0 < KTOT; k0 += 32) {
    // ---- stage A tile ----
    if (HAS_A0 && k0 < K0) {                       // bf16 source
      int m  = tid >> 1;                           // 0..127
      int c0 = (tid & 1) * 16;
      const unsigned short* src = A0 + (size_t)(m0 + m) * 256 + k0 + c0;
#pragma unroll
      for (int q = 0; q < 2; ++q) {
        U16x8 u; u.q = *(const uint4*)(src + q * 8);
#pragma unroll
        for (int j = 0; j < 8; ++j) lA[m * LDA + c0 + q * 8 + j] = u.s[j];
      }
    } else {                                       // fp32 source
      int kk = k0 - K0;
#pragma unroll
      for (int p = 0; p < 4; ++p) {
        int m  = p * 32 + (tid >> 3);
        int cg = (tid & 7) * 4;
        const float4 v = *(const float4*)(A1 + (size_t)(m0 + m) * 256 + kk + cg);
        int o = m * LDA + cg;
        lA[o + 0] = f2bf(v.x); lA[o + 1] = f2bf(v.y);
        lA[o + 2] = f2bf(v.z); lA[o + 3] = f2bf(v.w);
      }
    }
    // ---- stage W tile (32 x 256 fp32 -> bf16, transposed to [n][k]) ----
    {
      int k  = tid >> 3;
      int c0 = (tid & 7) * 32;
      const float* src = W + (size_t)(k0 + k) * 256 + c0;
#pragma unroll
      for (int q = 0; q < 8; ++q) {
        float4 v = ((const float4*)src)[q];
        int c = c0 + q * 4;
        lB[(c + 0) * LDA + k] = f2bf(v.x);
        lB[(c + 1) * LDA + k] = f2bf(v.y);
        lB[(c + 2) * LDA + k] = f2bf(v.z);
        lB[(c + 3) * LDA + k] = f2bf(v.w);
      }
    }
    __syncthreads();

    FragBF a[2], bfr[8];
#pragma unroll
    for (int mt = 0; mt < 2; ++mt) {
      int aOff = (wm * 32 + mt * 16 + l16) * LDA;
      a[mt].q[0] = *(const uint4*)&lA[aOff + 8 * hx];
      a[mt].q[1] = *(const uint4*)&lA[aOff + 16 + 8 * hx];
    }
#pragma unroll
    for (int nt = 0; nt < 8; ++nt) {
      int bOff = (wn * 128 + nt * 16 + l16) * LDA;
      bfr[nt].q[0] = *(const uint4*)&lB[bOff + 8 * hx];
      bfr[nt].q[1] = *(const uint4*)&lB[bOff + 16 + 8 * hx];
    }
#pragma unroll
    for (int mt = 0; mt < 2; ++mt)
#pragma unroll
      for (int nt = 0; nt < 8; ++nt)
        acc[mt][nt] = __builtin_amdgcn_wmma_f32_16x16x32_bf16(
            false, a[mt].v, false, bfr[nt].v, (short)0, acc[mt][nt], false, false);
    __syncthreads();
  }

  // ---- epilogue: bias + relu ----
#pragma unroll
  for (int mt = 0; mt < 2; ++mt) {
#pragma unroll
    for (int nt = 0; nt < 8; ++nt) {
      int gn = wn * 128 + nt * 16 + l16;
      float bn = bias[gn];
#pragma unroll
      for (int r = 0; r < 8; ++r) {
        int ml = wm * 32 + mt * 16 + r + 8 * hx;
        float v = acc[mt][nt][r] + bn;
        v = fmaxf(v, 0.f);
        if constexpr (OUT_F32)  OutF [(size_t)(m0 + ml) * 256 + gn] = v;
        if constexpr (OUT_BF16) OutBf[(size_t)(m0 + ml) * 256 + gn] = f2bf(v);
      }
    }
  }
}

// ---------------------------------------------------------------------------
extern "C" void kernel_launch(void* const* d_in, const int* in_sizes, int n_in,
                              void* d_out, int out_size, void* d_ws, size_t ws_size,
                              hipStream_t stream) {
  (void)in_sizes; (void)n_in; (void)out_size; (void)ws_size;

  const float* H    = (const float*)d_in[0];
  const float* E    = (const float*)d_in[1];
  const float* x    = (const float*)d_in[2];
  const float* W_ne = (const float*)d_in[3];
  const float* b_ne = (const float*)d_in[4];
  const float* W_e  = (const float*)d_in[5];
  const float* b_e  = (const float*)d_in[6];
  const float* W_n  = (const float*)d_in[7];
  const float* b_n  = (const float*)d_in[8];

  const size_t MFD = (size_t)BATCH * 2048 * FD;     // 8,388,608 elements

  char* ws = (char*)d_ws;
  float* colsum          = (float*)ws;          ws += (size_t)BATCH * NE * 4;
  float* sNodeInv        = (float*)ws;          ws += (size_t)BATCH * NE * 4;
  float* sEdgeInv        = (float*)ws;          ws += (size_t)BATCH * NN * 4;
  unsigned short* neBf   = (unsigned short*)ws; ws += MFD * 2;
  unsigned short* xEBf   = (unsigned short*)ws; ws += MFD * 2;
  unsigned short* newEBf = (unsigned short*)ws; ws += MFD * 2;
  unsigned short* ExBf   = (unsigned short*)ws; ws += MFD * 2;

  float* outE = (float*)d_out;        // new_E: (16,2048,256)
  float* outX = outE + MFD;           // new_x: (16,2048,256)

  // 1) normalization sums
  hipMemsetAsync(colsum, 0, (size_t)BATCH * NE * 4, stream);
  sums_kernel<<<dim3(16, 16), 256, 0, stream>>>(H, colsum, sEdgeInv);
  finalize_inv<<<(BATCH * NE + 255) / 256, 256, 0, stream>>>(colsum, sNodeInv,
                                                             BATCH * NE);
  // 2) ne = relu(x @ W_ne + b_ne)                    -> bf16
  lin_gemm<false, false, true, 256><<<dim3(256), 256, 0, stream>>>(
      nullptr, x, W_ne, b_ne, nullptr, neBf);
  // 3) x_to_E = Hnorm^T @ ne                         -> bf16
  h_gemm<true><<<dim3(16, 16), 256, 0, stream>>>(H, neBf, sNodeInv, xEBf);
  // 4) new_E = relu([x_to_E, E] @ W_e + b_e)         -> f32 out + bf16
  lin_gemm<true, true, true, 512><<<dim3(256), 256, 0, stream>>>(
      xEBf, E, W_e, b_e, outE, newEBf);
  // 5) E_to_x = Hnorm @ new_E                        -> bf16
  h_gemm<false><<<dim3(16, 16), 256, 0, stream>>>(H, newEBf, sEdgeInv, ExBf);
  // 6) new_x = relu([E_to_x, x] @ W_n + b_n)         -> f32 out
  lin_gemm<true, true, false, 512><<<dim3(256), 256, 0, stream>>>(
      ExBf, x, W_n, b_n, outX, nullptr);
}
